// CausalMultiheadAttention_82738249990828
// MI455X (gfx1250) — matrix operand order repaired and checked by reading, not compiled, not run on hardware
//
#include <hip/hip_runtime.h>
#include <hip/hip_bf16.h>

#define HD __device__ __forceinline__

typedef __bf16 bf16;
typedef __attribute__((ext_vector_type(16))) __bf16 v16bf;
typedef __attribute__((ext_vector_type(8)))  __bf16 v8bf;
typedef __attribute__((ext_vector_type(8)))  float  v8f;

static constexpr int NH  = 16;
static constexpr int EMB = 1024;
static constexpr int DK  = 64;
static constexpr int DV  = 64;
static constexpr int Bz  = 4;
static constexpr int Lsq = 2048;
static constexpr int Mrows = Bz * Lsq;   // 8192

HD v16bf mk16(v8bf lo, v8bf hi) {
  return __builtin_shufflevector(lo, hi, 0,1,2,3,4,5,6,7,8,9,10,11,12,13,14,15);
}
HD v8f wmma_bf16(v16bf a, v16bf b, v8f c) {
  return __builtin_amdgcn_wmma_f32_16x16x32_bf16(false, a, false, b, (short)0, c,
                                                 false, false);
}
HD v8bf ldsv8(const bf16* p) { return *(const v8bf*)p; }

// ---- DPP 16-lane butterfly reductions (VALU, no LDS traffic) --------------
// xor masks {1,2,7,15} span lane bits 0..3 -> full reduction per 16-lane half.
template <int CTRL>
HD float dppf(float x) {
  return __int_as_float(
      __builtin_amdgcn_update_dpp(0, __float_as_int(x), CTRL, 0xf, 0xf, true));
}
HD float rmax16(float v) {
  v = fmaxf(v, dppf<0x0B1>(v));   // quad_perm [1,0,3,2]  (xor 1)
  v = fmaxf(v, dppf<0x04E>(v));   // quad_perm [2,3,0,1]  (xor 2)
  v = fmaxf(v, dppf<0x141>(v));   // ROW_HALF_MIRROR      (xor 7)
  v = fmaxf(v, dppf<0x140>(v));   // ROW_MIRROR           (xor 15)
  return v;
}
HD float rsum16(float v) {
  v += dppf<0x0B1>(v);
  v += dppf<0x04E>(v);
  v += dppf<0x141>(v);
  v += dppf<0x140>(v);
  return v;
}

// ---- async global -> LDS copy (CDNA5, tracked with ASYNCcnt) --------------
HD void async_b128(unsigned lds_byte_addr, unsigned gvoff, const void* sbase) {
  asm volatile("global_load_async_to_lds_b128 %0, %1, %2"
               :: "v"(lds_byte_addr), "v"(gvoff),
                  "s"((unsigned long long)(uintptr_t)sbase)
               : "memory");
}
HD void wait_async0() { asm volatile("s_wait_asynccnt 0" ::: "memory"); }
HD unsigned lds_off(const void* p) { return (unsigned)(uintptr_t)p; }

// ---------------------------------------------------------------------------
// C[M x N] = X[M x K] * W[K x N].  64x128 tile per 128-thread block; each
// wave owns a 64x32 strip (8 accumulators), A-fragments reused across the two
// B-fragments: 8 WMMAs per 32-wide K step.
// ---------------------------------------------------------------------------
template <typename XT, typename OT>
__global__ __launch_bounds__(128)
void gemm64x128(const XT* __restrict__ X, const float* __restrict__ W,
                OT* __restrict__ Y, int M, int K, int N) {
  __shared__ __align__(16) bf16 Xs[64][32];    // [row][k]
  __shared__ __align__(16) bf16 Wt[128][32];   // [n][k]  (transposed)

  const int tid  = threadIdx.x;
  const int wv   = tid >> 5;
  const int ln   = tid & 31;
  const int half = ln >> 4;
  const int m    = ln & 15;
  const int tM   = blockIdx.y * 64;
  const int tN   = blockIdx.x * 128;

  v8f acc[4][2] = {};

  for (int k0 = 0; k0 < K; k0 += 32) {
    if (k0 + 32 < K) {                        // hint next K-slice into caches
      __builtin_prefetch(&X[(size_t)(tM + (tid >> 1)) * K + k0 + 32], 0, 0);
      __builtin_prefetch(&W[(size_t)(k0 + 32 + (tid & 31)) * N + tN], 0, 0);
    }
    __syncthreads();
#pragma unroll
    for (int e = tid; e < 64 * 32; e += 128) {      // X tile
      int r = e >> 5, c = e & 31;
      Xs[r][c] = (bf16)(float)X[(size_t)(tM + r) * K + k0 + c];
    }
#pragma unroll
    for (int e = tid; e < 32 * 128; e += 128) {     // W tile -> transposed
      int kk = e >> 7, n = e & 127;
      Wt[n][kk] = (bf16)W[(size_t)(k0 + kk) * N + tN + n];
    }
    __syncthreads();

    v16bf bfr[2];
#pragma unroll
    for (int j = 0; j < 2; ++j)                     // B frag: k = half*8 + e, 16 + half*8 + e
      bfr[j] = mk16(ldsv8(&Wt[wv * 32 + j * 16 + m][half * 8]),
                    ldsv8(&Wt[wv * 32 + j * 16 + m][16 + half * 8]));
#pragma unroll
    for (int r = 0; r < 4; ++r) {
      v16bf afr = mk16(ldsv8(&Xs[r * 16 + m][half * 8]),
                       ldsv8(&Xs[r * 16 + m][16 + half * 8]));
#pragma unroll
      for (int j = 0; j < 2; ++j)
        acc[r][j] = wmma_bf16(afr, bfr[j], acc[r][j]);
    }
  }

#pragma unroll
  for (int r = 0; r < 4; ++r)
#pragma unroll
    for (int j = 0; j < 2; ++j)
#pragma unroll
      for (int i = 0; i < 8; ++i) {
        int row = tM + r * 16 + half * 8 + i;
        int col = tN + wv * 32 + j * 16 + m;
        Y[(size_t)row * N + col] = (OT)acc[r][j][i];
      }
}

// ---------------------------------------------------------------------------
// Flash attention: one block per (b, h, 64-row q block).  Q and K tiles are
// staged with async global->LDS copies (ASYNCcnt); V goes through VGPRs since
// it needs a transpose for the B-fragment layout.
// ---------------------------------------------------------------------------
__global__ __launch_bounds__(128)
void attn_fwd(const bf16* __restrict__ Q, const bf16* __restrict__ Kp,
              const bf16* __restrict__ Vp, const int* __restrict__ pad,
              bf16* __restrict__ O) {
  const int qb = blockIdx.x, h = blockIdx.y, b = blockIdx.z;
  const int tid = threadIdx.x;
  const int wv  = tid >> 5;
  const int ln  = tid & 31;
  const int half = ln >> 4;
  const int m    = ln & 15;
  const int rowbytes = NH * DK * 2;     // 2048B stride between seq positions

  __shared__ __align__(16) bf16 Qs[64][64];      // [qrow][dk]
  __shared__ __align__(16) bf16 Ks[32][64];      // [key][dk]
  __shared__ __align__(16) bf16 Vt[64][32];      // [vdim][key]
  __shared__ __align__(16) bf16 Ps[4][16][32];   // per-wave P staging
  __shared__ float kbias[32];

  // ---- async-stage the whole Q tile (8 KB, 4 x b128 per thread) ----
  {
    const bf16* qbase = Q + (size_t)(b * Lsq + qb * 64) * (NH * DK) + h * DK;
    unsigned ql = lds_off(&Qs[0][0]);
#pragma unroll
    for (int it = 0; it < 4; ++it) {
      unsigned e = (unsigned)tid * 16 + it * 2048;   // byte index in tile
      unsigned r = e >> 7, c = e & 127;
      async_b128(ql + e, r * rowbytes + c, qbase);
    }
  }
  wait_async0();
  __syncthreads();

  v16bf qf[2];
#pragma unroll
  for (int c2 = 0; c2 < 2; ++c2)
    qf[c2] = mk16(ldsv8(&Qs[wv * 16 + m][c2 * 32 + half * 8]),
                  ldsv8(&Qs[wv * 16 + m][c2 * 32 + 16 + half * 8]));

  float mst[8], sst[8];
#pragma unroll
  for (int i = 0; i < 8; ++i) { mst[i] = -1e30f; sst[i] = 0.f; }
  v8f ov[4] = {};

  const int nkb = (qb + 1) * 2;                   // causal: keys < (qb+1)*64
  for (int kb = 0; kb < nkb; ++kb) {
    __syncthreads();                              // all waves done with tiles
    // K tile: async copy (4 KB, 2 x b128 per thread)
    {
      const bf16* kbase = Kp + (size_t)(b * Lsq + kb * 32) * (NH * DK) + h * DK;
      unsigned kl = lds_off(&Ks[0][0]);
#pragma unroll
      for (int it = 0; it < 2; ++it) {
        unsigned e = (unsigned)tid * 16 + it * 2048;
        unsigned r = e >> 7, c = e & 127;
        async_b128(kl + e, r * rowbytes + c, kbase);
      }
    }
    // V tile: VGPR path with transpose into Vt[n][k]
    for (int e = tid; e < 32 * 64; e += 128) {
      int r = e >> 6, c = e & 63;
      Vt[c][r] = Vp[(size_t)(b * Lsq + kb * 32 + r) * (NH * DV) + h * DV + c];
    }
    if (tid < 32) kbias[tid] = pad[b * Lsq + kb * 32 + tid] ? 0.f : -1e30f;
    wait_async0();
    __syncthreads();

    // S = Q * K^T : 2 n-tiles x 2 k-chunks
    v8f s[2] = {};
#pragma unroll
    for (int t = 0; t < 2; ++t)
#pragma unroll
      for (int c2 = 0; c2 < 2; ++c2) {
        v16bf kf = mk16(ldsv8(&Ks[t * 16 + m][c2 * 32 + half * 8]),
                        ldsv8(&Ks[t * 16 + m][c2 * 32 + 16 + half * 8]));
        s[t] = wmma_bf16(qf[c2], kf, s[t]);
      }

    // scale 1/sqrt(dk), causal mask, pad mask
    const int qrow0 = qb * 64 + wv * 16 + half * 8;
#pragma unroll
    for (int t = 0; t < 2; ++t) {
      const int kcol = kb * 32 + t * 16 + m;
      const float bias = kbias[t * 16 + m];
#pragma unroll
      for (int i = 0; i < 8; ++i) {
        float v = s[t][i] * 0.125f + bias;
        s[t][i] = (kcol <= qrow0 + i) ? v : -1e30f;
      }
    }

    // online softmax update; reductions via DPP butterflies
#pragma unroll
    for (int i = 0; i < 8; ++i) {
      float rm = rmax16(fmaxf(s[0][i], s[1][i]));
      float mn = fmaxf(mst[i], rm);
      float f  = __expf(mst[i] - mn);
      mst[i] = mn;
      float p0 = __expf(s[0][i] - mn);
      float p1 = __expf(s[1][i] - mn);
      s[0][i] = p0; s[1][i] = p1;
      sst[i] = sst[i] * f + rsum16(p0 + p1);
#pragma unroll
      for (int tt = 0; tt < 4; ++tt) ov[tt][i] *= f;
    }

    // C-layout P -> bf16 A-layout via per-wave LDS staging
#pragma unroll
    for (int t = 0; t < 2; ++t)
#pragma unroll
      for (int i = 0; i < 8; ++i)
        Ps[wv][half * 8 + i][t * 16 + m] = (bf16)s[t][i];
    __builtin_amdgcn_wave_barrier();
    asm volatile("s_wait_dscnt 0" ::: "memory");   // same-wave LDS RAW

    v16bf pf = mk16(ldsv8(&Ps[wv][m][half * 8]),
                    ldsv8(&Ps[wv][m][16 + half * 8]));
#pragma unroll
    for (int tt = 0; tt < 4; ++tt) {
      v16bf vf = mk16(ldsv8(&Vt[tt * 16 + m][half * 8]),
                      ldsv8(&Vt[tt * 16 + m][16 + half * 8]));
      ov[tt] = wmma_bf16(pf, vf, ov[tt]);
    }
  }

  // normalize and write attn output (B, L, H*DV) as bf16
#pragma unroll
  for (int i = 0; i < 8; ++i) {
    float inv = 1.f / sst[i];
    int row = qb * 64 + wv * 16 + half * 8 + i;
#pragma unroll
    for (int tt = 0; tt < 4; ++tt)
      O[(size_t)(b * Lsq + row) * (NH * DV) + h * DV + tt * 16 + m] =
          (bf16)(ov[tt][i] * inv);
  }
}

// ---------------------------------------------------------------------------
extern "C" void kernel_launch(void* const* d_in, const int* in_sizes, int n_in,
                              void* d_out, int out_size, void* d_ws, size_t ws_size,
                              hipStream_t stream) {
  (void)in_sizes; (void)n_in; (void)out_size; (void)ws_size;
  const float* query = (const float*)d_in[0];
  const float* key   = (const float*)d_in[1];
  const float* value = (const float*)d_in[2];
  const int*   pad   = (const int*)d_in[3];
  const float* Wq    = (const float*)d_in[4];
  const float* Wk    = (const float*)d_in[5];
  const float* Wv    = (const float*)d_in[6];
  const float* Wo    = (const float*)d_in[7];
  float* out = (float*)d_out;

  const size_t tile = (size_t)Mrows * EMB;   // 8M elements (16 MB bf16) each
  bf16* qp = (bf16*)d_ws;
  bf16* kp = qp + tile;
  bf16* vp = kp + tile;
  bf16* ap = vp + tile;

  dim3 blk(128);
  dim3 gg(EMB / 128, Mrows / 64);

  hipLaunchKernelGGL((gemm64x128<float, bf16>), gg, blk, 0, stream,
                     query, Wq, qp, Mrows, EMB, EMB);
  hipLaunchKernelGGL((gemm64x128<float, bf16>), gg, blk, 0, stream,
                     key, Wk, kp, Mrows, EMB, EMB);
  hipLaunchKernelGGL((gemm64x128<float, bf16>), gg, blk, 0, stream,
                     value, Wv, vp, Mrows, EMB, EMB);

  dim3 ga(Lsq / 64, NH, Bz);
  hipLaunchKernelGGL(attn_fwd, ga, blk, 0, stream, qp, kp, vp, pad, ap);

  hipLaunchKernelGGL((gemm64x128<bf16, float>), gg, blk, 0, stream,
                     ap, Wo, out, Mrows, EMB, EMB);
}
